// GeneralStandardAttention_87789131530846
// MI455X (gfx1250) — matrix operand order repaired and checked by reading, not compiled, not run on hardware
//
#include <hip/hip_runtime.h>

// ---------------------------------------------------------------------------
// GeneralStandardAttention for MI455X (gfx1250, wave32, WMMA)
// B=2, N=2048, C=1024, H=16, D=64.  Memory-bound on att_weights (537MB fp32):
// minimize passes over it; all GEMMs via v_wmma_f32_16x16x32_f16 (f32 accum).
// Q/K tile staging uses GLOBAL_LOAD_ASYNC_TO_LDS_B128 (ASYNCcnt path).
// ---------------------------------------------------------------------------

typedef __attribute__((ext_vector_type(16))) _Float16 v16h;
typedef __attribute__((ext_vector_type(8)))  float    v8f;
typedef __attribute__((ext_vector_type(4)))  _Float16 v4h;

#define NB 2048
#define CB 1024
#define HH 16
#define DD 64
#define MASKV (-10000.0f)
#define SCALE 0.125f   // D^-0.5 = 64^-0.5

// Two contiguous 16B chunks -> one 32B (16 x f16) WMMA fragment. Both chunks
// are 16B-aligned in LDS, so these lower to ds_load_b128.
__device__ __forceinline__ v16h frag2(const _Float16* p0, const _Float16* p1) {
  union { v16h v; uint4 u[2]; } r;
  r.u[0] = *(const uint4*)p0;
  r.u[1] = *(const uint4*)p1;
  return r.v;
}

__device__ __forceinline__ v4h cvt4(float4 f) {
  v4h h;
  h[0] = (_Float16)f.x; h[1] = (_Float16)f.y;
  h[2] = (_Float16)f.z; h[3] = (_Float16)f.w;
  return h;
}

// Async 16B global -> LDS copy (no VGPR data round-trip, tracked by ASYNCcnt).
// On amdgcn the low 32 bits of a generic pointer to __shared__ are the LDS
// byte offset, which is what the VDST address operand wants.
__device__ __forceinline__ void async_copy_b128(void* lds_dst, const void* gsrc) {
  unsigned           dst = (unsigned)(unsigned long long)lds_dst;
  unsigned long long src = (unsigned long long)gsrc;
  asm volatile("global_load_async_to_lds_b128 %0, %1, off"
               :: "v"(dst), "v"(src) : "memory");
}
__device__ __forceinline__ void async_wait0() {
  asm volatile("s_wait_asynccnt 0" ::: "memory");
}

// ---------------------------------------------------------------------------
// Kernel 1: projections.  z = b*3 + p, p in {q,k,v}.
// Computes X @ W (X = hs or ehs, W = W_q or a half of W_c), 128x128 tile per
// block, K-loop over 1024 in steps of 32.  Output f16 in [b,h,n,64] layout.
// 8 waves: wave (wM,wN) in 2x4 grid covers 64 rows x 32 cols = 4x2 WMMA tiles.
// (f32 -> f16 conversion forces the load-to-VGPR path here.)
// ---------------------------------------------------------------------------
__global__ __launch_bounds__(256) void proj_gemm(
    const float* __restrict__ hs, const float* __restrict__ ehs,
    const float* __restrict__ Wq, const float* __restrict__ Wc,
    _Float16* __restrict__ qw, _Float16* __restrict__ kw,
    _Float16* __restrict__ vw)
{
  __shared__ _Float16 sA[128][40];  // [m][k], row stride 80B (16B multiple)
  __shared__ _Float16 sB[128][40];  // [col][k] (W tile transposed)

  const int tid  = threadIdx.x, lane = tid & 31, wv = tid >> 5;
  const int z = blockIdx.z, b = z / 3, p = z % 3;
  const int rbase = blockIdx.x * 128, cbase = blockIdx.y * 128;

  const float* A = (p == 0 ? hs : ehs) + (size_t)b * NB * CB;
  const float* W; int ldw, coff;
  if (p == 0) { W = Wq; ldw = CB;     coff = 0; }
  else        { W = Wc; ldw = 2 * CB; coff = (p == 2) ? CB : 0; }
  _Float16* outw = (p == 0) ? qw : (p == 1 ? kw : vw);

  const int waveM = wv & 1, waveN = wv >> 1;
  const int half = lane >> 4, m16 = lane & 15;

  v8f acc[4][2] = {};

  for (int kk = 0; kk < CB; kk += 32) {
    // A tile 128x32 f32 -> f16 LDS (4 x float4 per thread)
#pragma unroll
    for (int i = 0; i < 4; i++) {
      int ch = tid + i * 256;
      int r = ch >> 3, c4 = (ch & 7) << 2;
      float4 f = *(const float4*)(A + (size_t)(rbase + r) * CB + kk + c4);
      *(v4h*)&sA[r][c4] = cvt4(f);
    }
    // W tile 32x128 f32 -> transposed f16 LDS sB[col][k]
#pragma unroll
    for (int i = 0; i < 4; i++) {
      int ch = tid + i * 256;
      int kr = ch >> 5, c4 = (ch & 31) << 2;
      float4 f = *(const float4*)(W + (size_t)(kk + kr) * ldw + coff + cbase + c4);
      sB[c4 + 0][kr] = (_Float16)f.x;
      sB[c4 + 1][kr] = (_Float16)f.y;
      sB[c4 + 2][kr] = (_Float16)f.z;
      sB[c4 + 3][kr] = (_Float16)f.w;
    }
    if (kk + 32 < CB)  // global_prefetch_b8 for next A tile
      __builtin_prefetch(A + (size_t)(rbase + (tid >> 3)) * CB + kk + 32, 0, 0);
    __syncthreads();

    v16h af[4], bf[2];
#pragma unroll
    for (int t = 0; t < 4; t++) {
      int r = waveM * 64 + t * 16 + m16;
      af[t] = frag2(&sA[r][half * 8], &sA[r][16 + half * 8]);
    }
#pragma unroll
    for (int t = 0; t < 2; t++) {
      int c = waveN * 32 + t * 16 + m16;
      bf[t] = frag2(&sB[c][half * 8], &sB[c][16 + half * 8]);
    }
#pragma unroll
    for (int i = 0; i < 4; i++)
#pragma unroll
      for (int j = 0; j < 2; j++)
        acc[i][j] = __builtin_amdgcn_wmma_f32_16x16x32_f16(
            false, af[i], false, bf[j], (short)0, acc[i][j], false, false);
    __syncthreads();
  }

  // Epilogue: f32 accum -> f16, scatter into [b,h,n,64] (16 contiguous d per
  // half-wave).
#pragma unroll
  for (int i = 0; i < 4; i++)
#pragma unroll
    for (int j = 0; j < 2; j++)
#pragma unroll
      for (int g = 0; g < 8; g++) {
        int r = rbase + waveM * 64 + i * 16 + g + half * 8;
        int c = cbase + waveN * 32 + j * 16 + m16;
        int h = c >> 6, d = c & 63;
        outw[(((size_t)(b * HH + h)) * NB + r) * DD + d] = (_Float16)acc[i][j][g];
      }
}

// ---------------------------------------------------------------------------
// Kernel 2: S = scale * Q K^T, pad mask -> raw masked scores into att_weights
// region.  One 128x128 tile per block; K-dim = 64 (two K=32 WMMA steps).
// Q/K tiles are staged with global_load_async_to_lds_b128 (pure f16 copies,
// no conversion needed).  K-tile is naturally [n][k] in LDS = transposed (B)
// fragment layout -> ds_load_b128.
// ---------------------------------------------------------------------------
__global__ __launch_bounds__(256) void scores_kernel(
    const _Float16* __restrict__ qw, const _Float16* __restrict__ kw,
    const int* __restrict__ pad, float* __restrict__ attw)
{
  __shared__ _Float16 sQ[128][72];   // row stride 144B (16B multiple)
  __shared__ _Float16 sK[128][72];
  __shared__ unsigned char pr[128], pc[128];

  const int tid = threadIdx.x, lane = tid & 31, wv = tid >> 5;
  const int bh = blockIdx.z, b = bh >> 4;
  const int rbase = blockIdx.x * 128, cbase = blockIdx.y * 128;

#pragma unroll
  for (int i = 0; i < 4; i++) {
    int ch = tid + i * 256;
    int r = ch >> 3, c8 = (ch & 7) << 3;
    async_copy_b128(&sQ[r][c8], qw + ((size_t)bh * NB + rbase + r) * DD + c8);
    async_copy_b128(&sK[r][c8], kw + ((size_t)bh * NB + cbase + r) * DD + c8);
  }
  if (tid < 128) {
    pr[tid] = (unsigned char)(pad[b * NB + rbase + tid] != 0);
    pc[tid] = (unsigned char)(pad[b * NB + cbase + tid] != 0);
  }
  async_wait0();          // this wave's async copies have landed in LDS
  __syncthreads();        // all waves' copies visible

  const int waveM = wv & 1, waveN = wv >> 1;
  const int half = lane >> 4, m16 = lane & 15;

  v8f acc[4][2] = {};
#pragma unroll
  for (int s = 0; s < 2; s++) {
    v16h af[4], bf[2];
#pragma unroll
    for (int t = 0; t < 4; t++) {
      int r = waveM * 64 + t * 16 + m16;
      af[t] = frag2(&sQ[r][32 * s + half * 8], &sQ[r][32 * s + 16 + half * 8]);
    }
#pragma unroll
    for (int t = 0; t < 2; t++) {
      int c = waveN * 32 + t * 16 + m16;
      bf[t] = frag2(&sK[c][32 * s + half * 8], &sK[c][32 * s + 16 + half * 8]);
    }
#pragma unroll
    for (int i = 0; i < 4; i++)
#pragma unroll
      for (int j = 0; j < 2; j++)
        acc[i][j] = __builtin_amdgcn_wmma_f32_16x16x32_f16(
            false, af[i], false, bf[j], (short)0, acc[i][j], false, false);
  }

#pragma unroll
  for (int i = 0; i < 4; i++)
#pragma unroll
    for (int j = 0; j < 2; j++)
#pragma unroll
      for (int g = 0; g < 8; g++) {
        int r = waveM * 64 + i * 16 + g + half * 8;
        int c = waveN * 32 + j * 16 + m16;
        float v = acc[i][j][g] * SCALE;
        if (pr[r] | pc[c]) v = MASKV;
        attw[((size_t)bh * NB + rbase + r) * NB + cbase + c] = v;
      }
}

// ---------------------------------------------------------------------------
// Kernel 3: in-place row softmax.  One 256-thread block per row (2048 f32,
// 8 elements = 2 float4 per thread).  Subtracting the row max matches the
// reference exactly (softmax is shift-invariant).
// ---------------------------------------------------------------------------
__global__ __launch_bounds__(256) void softmax_kernel(float* __restrict__ attw)
{
  __shared__ float red[256];
  const int t = threadIdx.x;
  float* row = attw + (size_t)blockIdx.x * NB;
  const float4* rp = (const float4*)row;
  float4 a = rp[t * 2], c = rp[t * 2 + 1];

  float m = fmaxf(fmaxf(fmaxf(a.x, a.y), fmaxf(a.z, a.w)),
                  fmaxf(fmaxf(c.x, c.y), fmaxf(c.z, c.w)));
  red[t] = m; __syncthreads();
  for (int s = 128; s > 0; s >>= 1) {
    if (t < s) red[t] = fmaxf(red[t], red[t + s]);
    __syncthreads();
  }
  float rmax = red[0];
  __syncthreads();

  float e0 = __expf(a.x - rmax), e1 = __expf(a.y - rmax);
  float e2 = __expf(a.z - rmax), e3 = __expf(a.w - rmax);
  float e4 = __expf(c.x - rmax), e5 = __expf(c.y - rmax);
  float e6 = __expf(c.z - rmax), e7 = __expf(c.w - rmax);

  red[t] = (e0 + e1 + e2 + e3) + (e4 + e5 + e6 + e7);
  __syncthreads();
  for (int s = 128; s > 0; s >>= 1) {
    if (t < s) red[t] += red[t + s];
    __syncthreads();
  }
  float inv = 1.0f / red[0];

  ((float4*)row)[t * 2]     = make_float4(e0 * inv, e1 * inv, e2 * inv, e3 * inv);
  ((float4*)row)[t * 2 + 1] = make_float4(e4 * inv, e5 * inv, e6 * inv, e7 * inv);
}

// ---------------------------------------------------------------------------
// Kernel 4: att = P @ V per (b,h).  M=2048 (128 rows per block), N=64, K=2048.
// P read fp32 (likely L2-resident right after softmax), converted to f16 in
// LDS; V transposed into LDS so B-fragments are contiguous ds_load_b128.
// Output written with the (B,N,H,D) -> (B,N,C) interleave.
// ---------------------------------------------------------------------------
__global__ __launch_bounds__(256) void attv_kernel(
    const float* __restrict__ attw, const _Float16* __restrict__ vw,
    float* __restrict__ out)
{
  __shared__ _Float16 sP[128][40];   // [m][k]
  __shared__ _Float16 sVt[64][40];   // [d][k] (V tile transposed)

  const int tid = threadIdx.x, lane = tid & 31, wv = tid >> 5;
  const int bh = blockIdx.z, b = bh >> 4, h = bh & 15;
  const int rbase = blockIdx.x * 128;
  const int half = lane >> 4, m16 = lane & 15;

  v8f acc[4] = {};

  for (int kk = 0; kk < NB; kk += 32) {
#pragma unroll
    for (int i = 0; i < 4; i++) {
      int ch = tid + i * 256;
      int r = ch >> 3, c4 = (ch & 7) << 2;
      float4 f = *(const float4*)(attw + ((size_t)bh * NB + rbase + r) * NB + kk + c4);
      *(v4h*)&sP[r][c4] = cvt4(f);
    }
    {
      int kr = tid >> 3, d8 = (tid & 7) << 3;
      union { uint4 u; _Float16 hx[8]; } vld;
      vld.u = *(const uint4*)(vw + ((size_t)bh * NB + kk + kr) * DD + d8);
#pragma unroll
      for (int j = 0; j < 8; j++) sVt[d8 + j][kr] = vld.hx[j];
    }
    __syncthreads();

    int r = wv * 16 + m16;
    v16h af = frag2(&sP[r][half * 8], &sP[r][16 + half * 8]);
#pragma unroll
    for (int t = 0; t < 4; t++) {
      v16h bf = frag2(&sVt[t * 16 + m16][half * 8], &sVt[t * 16 + m16][16 + half * 8]);
      acc[t] = __builtin_amdgcn_wmma_f32_16x16x32_f16(
          false, af, false, bf, (short)0, acc[t], false, false);
    }
    __syncthreads();
  }

#pragma unroll
  for (int t = 0; t < 4; t++)
#pragma unroll
    for (int g = 0; g < 8; g++) {
      int r = rbase + wv * 16 + g + half * 8;
      int d = t * 16 + m16;
      out[((size_t)b * NB + r) * CB + h * DD + d] = acc[t][g];
    }
}

// ---------------------------------------------------------------------------
extern "C" void kernel_launch(void* const* d_in, const int* in_sizes, int n_in,
                              void* d_out, int out_size, void* d_ws, size_t ws_size,
                              hipStream_t stream) {
  const float* hs  = (const float*)d_in[0];
  const float* ehs = (const float*)d_in[1];
  const int*   pad = (const int*)d_in[2];
  const float* Wq  = (const float*)d_in[3];
  const float* Wc  = (const float*)d_in[4];

  float* out  = (float*)d_out;                       // att: B*N*C = 4,194,304 f32
  float* attw = out + (size_t)2 * NB * CB;           // att_weights: B*H*N*N f32

  const size_t per = (size_t)2 * HH * NB * DD;       // 4,194,304 f16 each
  _Float16* qw = (_Float16*)d_ws;
  _Float16* kw = qw + per;
  _Float16* vw = kw + per;                           // total 24 MB of d_ws

  // 1) q/k/v projections (WMMA f16, f32 accum) -> ws
  proj_gemm<<<dim3(16, 8, 6), 256, 0, stream>>>(hs, ehs, Wq, Wc, qw, kw, vw);
  // 2) raw masked scores -> att_weights region of d_out (async-LDS staging)
  scores_kernel<<<dim3(16, 16, 32), 256, 0, stream>>>(qw, kw, pad, attw);
  // 3) in-place row softmax over att_weights
  softmax_kernel<<<2 * HH * NB, 256, 0, stream>>>(attw);
  // 4) att = P @ V -> att region of d_out
  attv_kernel<<<dim3(16, 1, 32), 256, 0, stream>>>(attw, vw, out);
}
